// GemLiteLinearTriton_55319178773088
// MI455X (gfx1250) — matrix operand order repaired and checked by reading, not compiled, not run on hardware
//
#include <hip/hip_runtime.h>

// ---------------------------------------------------------------------------
// W4A32 grouped-quant GEMM for gfx1250 (MI455X).
//   out[M,N] = x[M,K] @ ((W4 - zeros) * scales) + bias
// f16 WMMA (v_wmma_f32_16x16x32_f16) with f32 accumulate.
// A pre-converted to f16 (d_ws) and streamed global->LDS with async copies;
// W dequantized with packed v_perm/v_pk_fma f16 math.
// ---------------------------------------------------------------------------

typedef _Float16 h16 __attribute__((ext_vector_type(16)));
typedef _Float16 h8  __attribute__((ext_vector_type(8)));
typedef _Float16 h4  __attribute__((ext_vector_type(4)));
typedef _Float16 h2  __attribute__((ext_vector_type(2)));
typedef float    v8f __attribute__((ext_vector_type(8)));
typedef float    f4  __attribute__((ext_vector_type(4)));
typedef int      g4i __attribute__((__vector_size__(16)));   // matches builtin param

#define MDIM 4096
#define NDIM 4096
#define KDIM 4096
#define QGROUP 128           // scales/zeros group size along K
#define BTM 256              // block tile M (tall: amortizes B dequant)
#define BTN 128              // block tile N
#define KT 32                // K tile = one v_wmma_f32_16x16x32_f16 step
#define LDA 40               // LDS row stride in halves (32 + 8 pad, 16B-aligned rows)
#define NTHREADS 256         // 8 waves (wave32)

#if defined(__gfx1250__) && __has_builtin(__builtin_amdgcn_global_load_async_to_lds_b128)
#define HAVE_ASYNC_LDS 1
#else
#define HAVE_ASYNC_LDS 0
#endif

union HV { h16 v; h8 h[2]; };

static __device__ __forceinline__ void waitAsync() {
#if HAVE_ASYNC_LDS
#if __has_builtin(__builtin_amdgcn_s_wait_asynccnt)
    __builtin_amdgcn_s_wait_asynccnt(0);
#else
    asm volatile("s_wait_asynccnt 0" ::: "memory");
#endif
#endif
}

// Assemble nibble pair (2j, 2j+1) of q into a packed half2 {1024+w_lo, 1024+w_hi}.
static __device__ __forceinline__ unsigned nib2half(unsigned e, unsigned o, int j) {
#if __has_builtin(__builtin_amdgcn_perm)
    // bytes of result: [e.byte(j), 0x00, o.byte(j), 0x00]; sel 4+j->src0, j->src1, 0x0C->0
    unsigned sel = 0x0C000C00u | ((unsigned)j << 16) | (unsigned)(4 + j);
    return __builtin_amdgcn_perm(e, o, sel) | 0x64006400u;
#else
    unsigned lo = (e >> (8 * j)) & 0xFu;
    unsigned hi = (o >> (8 * j)) & 0xFu;
    return 0x64006400u | lo | (hi << 16);
#endif
}

// Dequantize one packed int32 (8 nibbles along K) to 8 halves: (w - z) * s.
static __device__ __forceinline__ h8 dequant8(int q, float s, float z) {
    unsigned uq = (unsigned)q;
    unsigned e  = uq & 0x0F0F0F0Fu;          // even nibbles (k = 0,2,4,6)
    unsigned o  = (uq >> 4) & 0x0F0F0F0Fu;   // odd  nibbles (k = 1,3,5,7)
    _Float16 sh = (_Float16)s;
    _Float16 zb = (_Float16)(-(z * (float)sh));
    h2 s2 = {sh, sh};
    h2 z2 = {zb, zb};
    h2 kf = {(_Float16)-1024.0f, (_Float16)-1024.0f};
    h8 r;
    #pragma unroll
    for (int j = 0; j < 4; ++j) {
        h2 hv = __builtin_bit_cast(h2, nib2half(e, o, j)); // {1024+w0, 1024+w1}
        h2 w  = hv + kf;                                   // exact {w0, w1} (v_pk_add_f16)
        h2 v  = __builtin_elementwise_fma(w, s2, z2);      // v_pk_fma_f16
        r[2 * j]     = v[0];
        r[2 * j + 1] = v[1];
    }
    return r;
}

// ---------------------------------------------------------------------------
// Main GEMM. AF16 = A already converted to f16 in xh (workspace).
// ---------------------------------------------------------------------------
template <bool AF16>
__global__ __launch_bounds__(NTHREADS)
void w4a32_gemm_wmma(const float*    __restrict__ x,
                     const _Float16* __restrict__ xh,
                     const int*      __restrict__ wq,      // [K/8][N]
                     const float*    __restrict__ scales,  // [K/128][N]
                     const float*    __restrict__ zeros,   // [K/128][N]
                     const float*    __restrict__ bias,    // [N]
                     float*          __restrict__ out)     // [M][N]
{
    __shared__ __align__(16) _Float16 As[2][BTM * LDA];    // A tile [row][k]
    __shared__ __align__(16) _Float16 Bs[2][BTN * LDA];    // W tile [n][k]

    const int tid   = threadIdx.x;
    const int lane  = tid & 31;
    const int l16   = lane & 15;
    const int lHi   = lane >> 4;          // 0: lanes 0-15, 1: lanes 16-31
    const int wave  = tid >> 5;           // 0..7
    const int wmOff = (wave >> 1) * 64;   // 4 waves along M, 64 rows each
    const int wnOff = (wave & 1) * 64;    // 2 waves along N, 64 cols each

    const int rowBase = blockIdx.y * BTM;
    const int colBase = blockIdx.x * BTN;

    // 4 (M-sub) x 4 (N-sub) accumulator tiles of 16x16 f32
    v8f acc[4][4];
    #pragma unroll
    for (int m = 0; m < 4; ++m)
        #pragma unroll
        for (int n = 0; n < 4; ++n)
            acc[m][n] = (v8f)0.0f;

    // Staging registers (unused paths are dead-code eliminated)
    f4    aS[8];                // AF16=false: 256x32 f32 tile -> 8 float4/thread
    h8    aH[4];                // AF16=true, no async: 4x b128/thread
    int   qStage[2];
    float sStage[2], zStage[2];

    // ---- A tile: global -> (LDS | regs) ----------------------------------
    auto issueA = [&](int kt, int buf) {
        const int kBase = kt * KT;
        if constexpr (AF16) {
            #pragma unroll
            for (int i = 0; i < 4; ++i) {               // 1024 16B chunks / tile
                int idx = tid + i * NTHREADS;
                int r = idx >> 2, c = idx & 3;
                const _Float16* g = xh + (size_t)(rowBase + r) * KDIM + kBase + c * 8;
#if HAVE_ASYNC_LDS
                unsigned lofs = (unsigned)(uintptr_t)(&As[buf][r * LDA + c * 8]);
                __builtin_amdgcn_global_load_async_to_lds_b128(
                    (__attribute__((address_space(1))) g4i*)(uintptr_t)g,
                    (__attribute__((address_space(3))) g4i*)(uintptr_t)lofs,
                    0, 0);
#else
                aH[i] = *(const h8*)g;
#endif
            }
        } else {
            #pragma unroll
            for (int i = 0; i < 8; ++i) {               // 2048 float4 / tile
                int idx = tid + i * NTHREADS;
                int r = idx >> 3, kq = idx & 7;
                aS[i] = *(const f4*)(x + (size_t)(rowBase + r) * KDIM + kBase + kq * 4);
            }
        }
        (void)buf;
    };

    auto storeA = [&](int buf) {
        if constexpr (AF16) {
#if !HAVE_ASYNC_LDS
            #pragma unroll
            for (int i = 0; i < 4; ++i) {
                int idx = tid + i * NTHREADS;
                int r = idx >> 2, c = idx & 3;
                *(h8*)(&As[buf][r * LDA + c * 8]) = aH[i];
            }
#endif
        } else {
            #pragma unroll
            for (int i = 0; i < 8; ++i) {
                int idx = tid + i * NTHREADS;
                int r = idx >> 3, kq = idx & 7;
                h4 t;
                #pragma unroll
                for (int j = 0; j < 4; ++j) t[j] = (_Float16)aS[i][j];
                *(h4*)(&As[buf][r * LDA + kq * 4]) = t;
            }
        }
        (void)buf;
    };

    // ---- W tile: global -> regs, dequant -> LDS --------------------------
    auto loadW = [&](int kt) {
        const int kBase = kt * KT;
        const int g = kBase / QGROUP;
        #pragma unroll
        for (int i = 0; i < 2; ++i) {                   // 4x128 packed ints
            int idx = tid + i * NTHREADS;
            int kp = idx >> 7, n = idx & 127;
            qStage[i] = wq    [(size_t)(kBase / 8 + kp) * NDIM + colBase + n];
            sStage[i] = scales[(size_t)g * NDIM + colBase + n];
            zStage[i] = zeros [(size_t)g * NDIM + colBase + n];
        }
    };

    auto storeW = [&](int buf) {
        #pragma unroll
        for (int i = 0; i < 2; ++i) {
            int idx = tid + i * NTHREADS;
            int kp = idx >> 7, n = idx & 127;
            h8 t = dequant8(qStage[i], sStage[i], zStage[i]);
            *(h8*)(&Bs[buf][n * LDA + kp * 8]) = t;     // 16B aligned
        }
    };

    // ---- LDS -> fragments -> 16x WMMA ------------------------------------
    auto compute = [&](int buf) {
        HV aF[4], bF[4];
        #pragma unroll
        for (int m = 0; m < 4; ++m) {
            // ISA 16-bit A 16x32 layout: lanes 0-15 hold K 0..7 & 16..23 of row l16,
            // lanes 16-31 hold K 8..15 & 24..31.
            const _Float16* p = &As[buf][(wmOff + m * 16 + l16) * LDA + lHi * 8];
            aF[m].h[0] = *(const h8*)p;
            aF[m].h[1] = *(const h8*)(p + 16);
        }
        #pragma unroll
        for (int n = 0; n < 4; ++n) {
            // B 32x16 layout (matches A): lanes 0-15 hold K 0..7 & 16..23 of col l16,
            // lanes 16-31 hold K 8..15 & 24..31.
            const _Float16* p = &Bs[buf][(wnOff + n * 16 + l16) * LDA + lHi * 8];
            bF[n].h[0] = *(const h8*)p;
            bF[n].h[1] = *(const h8*)(p + 16);
        }
        #pragma unroll
        for (int m = 0; m < 4; ++m)
            #pragma unroll
            for (int n = 0; n < 4; ++n)
                acc[m][n] = __builtin_amdgcn_wmma_f32_16x16x32_f16(
                    false, aF[m].v, false, bF[n].v,
                    (short)0, acc[m][n], false, false);
    };

    const int NT = KDIM / KT;   // 128 K-tiles

    // Prologue: fill buffer 0
    issueA(0, 0);
    loadW(0);
    storeW(0);
    storeA(0);
    waitAsync();
    __syncthreads();

    for (int kt = 0; kt < NT; ++kt) {
        const int cur = kt & 1;
        const bool hasNext = (kt + 1) < NT;

        if (hasNext) {
            issueA(kt + 1, cur ^ 1);                    // async A -> LDS[next]
            loadW(kt + 1);                              // W ints/scales -> regs
        }
        if (kt + 2 < NT) {                              // L2 prefetch packed W
            const int* p = wq + (size_t)((kt + 2) * (KT / 8) + (tid >> 7)) * NDIM
                              + colBase + (tid & 127);
            __builtin_prefetch(p, 0, 1);
        }

        compute(cur);                                   // 16x v_wmma

        if (hasNext) {
            storeW(cur ^ 1);                            // dequant -> LDS[next]
            storeA(cur ^ 1);                            // no-op on async path
        }
        waitAsync();                                    // async A complete
        __syncthreads();
    }

    // ---- epilogue: bias + store ------------------------------------------
    // C/D layout: VGPR i -> row = base + i + 8*lHi, col = base + l16.
    #pragma unroll
    for (int m = 0; m < 4; ++m) {
        #pragma unroll
        for (int n = 0; n < 4; ++n) {
            const int col = colBase + wnOff + n * 16 + l16;
            const float bv = bias[col];
            #pragma unroll
            for (int i = 0; i < 8; ++i) {
                const int row = rowBase + wmOff + m * 16 + lHi * 8 + i;
                out[(size_t)row * NDIM + col] = acc[m][n][i] + bv;
            }
        }
    }
}

// ---------------------------------------------------------------------------
// Pre-pass: x (f32) -> xh (f16), row-major, one shot over 16M elements.
// ---------------------------------------------------------------------------
__global__ __launch_bounds__(256)
void cvt_a_f16(const float* __restrict__ x, _Float16* __restrict__ xh)
{
    size_t idx = ((size_t)blockIdx.x * 256 + threadIdx.x) * 8;
    f4 a = *(const f4*)(x + idx);
    f4 b = *(const f4*)(x + idx + 4);
    h8 t;
    #pragma unroll
    for (int j = 0; j < 4; ++j) {
        t[j]     = (_Float16)a[j];
        t[4 + j] = (_Float16)b[j];
    }
    *(h8*)(xh + idx) = t;
}

extern "C" void kernel_launch(void* const* d_in, const int* in_sizes, int n_in,
                              void* d_out, int out_size, void* d_ws, size_t ws_size,
                              hipStream_t stream) {
    const float* x      = (const float*)d_in[0];
    const int*   wq     = (const int*)  d_in[1];
    const float* scales = (const float*)d_in[2];
    const float* zeros  = (const float*)d_in[3];
    const float* bias   = (const float*)d_in[4];
    float* out = (float*)d_out;

    dim3 grid(NDIM / BTN, MDIM / BTM);   // 32 x 16 blocks
    dim3 block(NTHREADS);

    const size_t need = (size_t)MDIM * KDIM * sizeof(_Float16);   // 32 MB
    if (ws_size >= need) {
        _Float16* xh = (_Float16*)d_ws;
        cvt_a_f16<<<dim3((MDIM * (size_t)KDIM) / (256 * 8)), dim3(256), 0, stream>>>(x, xh);
        w4a32_gemm_wmma<true><<<grid, block, 0, stream>>>(x, xh, wq, scales, zeros, bias, out);
    } else {
        w4a32_gemm_wmma<false><<<grid, block, 0, stream>>>(x, nullptr, wq, scales, zeros, bias, out);
    }
}
